// FactorNetID16_LargerLevelLearnable220911_75703093559368
// MI455X (gfx1250) — matrix operand order repaired and checked by reading, not compiled, not run on hardware
//
#include <hip/hip_runtime.h>
#include <hip/hip_bf16.h>

// ---------------------------------------------------------------------------
// FactorNet forward for MI455X (gfx1250, wave32, WMMA).
// Pipeline: extraction (wave-per-row, LDS staging) -> BN stat reduce ->
// pointwise branch -> bf16 WMMA encoder GEMMs (v_wmma_f32_16x16x32_bf16,
// B staged in LDS via global_load_async_to_lds_b128, k-loop software
// pipelined, 2 M-tiles per wave) -> final linear.
// Input flattening assumed in setup_inputs() dict *insertion* order:
//   0:x 1:ln.g 2:ln.b 3:lin_w 4:lin_b then per d in (10,15,30) 26 leaves:
//   ext_g ext_b dw_k dw_b g1 b1 pw_w pw_b g2 b2 {enc_w,enc_b,enc_g,enc_bb}x4
// ---------------------------------------------------------------------------

#define BATCH 65536
#define T30 30
#define NMAX (BATCH * 3)         // d=10 -> W=3
#define EPSV 1e-5f

typedef __attribute__((ext_vector_type(16))) __bf16 v16bf;
typedef __attribute__((ext_vector_type(8)))  float  v8f;

struct Pairs { unsigned char iu[120], ju[120]; };
static constexpr Pairs make_pairs() {
  Pairs p{}; int k = 0;
  for (int i = 0; i < 16; i++)
    for (int j = i + 1; j < 16; j++) { p.iu[k] = (unsigned char)i; p.ju[k] = (unsigned char)j; k++; }
  return p;
}
__constant__ Pairs PAIRS = make_pairs();

__device__ __forceinline__ unsigned short f2bf(float f) {
  __hip_bfloat16 h = __float2bfloat16(f);
  union { __hip_bfloat16 h; unsigned short u; } cv; cv.h = h; return cv.u;
}
__device__ __forceinline__ float bf2f(unsigned short u) {
  union { unsigned short u; __hip_bfloat16 h; } cv; cv.u = u; return __bfloat162float(cv.h);
}

// A fragment per CDNA5 16-bit A 16x32 layout: lane<16 rows M=lane hold
// K = {0..7,16..23}; lanes>=16 rows M=lane-16 hold K = {8..15,24..31}.
__device__ __forceinline__ v16bf load_afrag(const unsigned short* lo, const unsigned short* hi) {
  union { unsigned short u[16]; v16bf v; } r;
#pragma unroll
  for (int i = 0; i < 8; i++) { r.u[i] = lo[i]; r.u[8 + i] = hi[i]; }
  return r.v;
}
__device__ __forceinline__ v16bf load_bfrag(const unsigned short* p) {
  union { unsigned short u[16]; v16bf v; } r;
#pragma unroll
  for (int i = 0; i < 16; i++) r.u[i] = p[i];
  return r.v;
}

// CDNA5 async copy global->LDS (tracked by ASYNCcnt).
__device__ __forceinline__ void async_g2l_b128(unsigned lds_off, const void* gptr) {
  asm volatile("global_load_async_to_lds_b128 %0, %1, off"
               :: "v"(lds_off), "v"(gptr) : "memory");
}
__device__ __forceinline__ void wait_asynccnt0() {
  asm volatile("s_wait_asynccnt 0x0" ::: "memory");
}

// ---------------------------------------------------------------------------
__global__ void zero_kernel(float* p, int n) {
  for (int i = blockIdx.x * blockDim.x + threadIdx.x; i < n; i += gridDim.x * blockDim.x) p[i] = 0.f;
}

// One wave per (batch, window) row. Lanes 0-15 own channels (lanes 16-31
// duplicate channel work, then share the 120 cov pairs, pair p on lane p % 32).
__global__ void extract_kernel(
    const float* __restrict__ x, const float* __restrict__ ln_g, const float* __restrict__ ln_b,
    const float* __restrict__ dw_k, const float* __restrict__ dw_b,
    int d, int W, int Nrows,
    unsigned short* __restrict__ A0,   // [Nrows, 352] bf16 raw feat (+zero pad 336..351)
    float* __restrict__ h1raw,         // [Nrows, 16]
    float* __restrict__ featS,         // S1[320] S2[320]
    float* __restrict__ h1S)           // S1[16]  S2[16]
{
  __shared__ float xcbuf[8][30][16];
  __shared__ float stdbuf[8][16];
  const int lane = threadIdx.x & 31;
  const int wv   = threadIdx.x >> 5;
  const int half = lane >> 4;
  const int c    = lane & 15;

  float sS[5] = {0,0,0,0,0}, sQ[5] = {0,0,0,0,0};
  float covS[4] = {0,0,0,0}, covQ[4] = {0,0,0,0}, corS[4] = {0,0,0,0}, corQ[4] = {0,0,0,0};
  float h1s = 0.f, h1q = 0.f;
  const float lng = ln_g[c], lnb = ln_b[c], dwb = dw_b[c];
  const float invd = 1.f / (float)d;
  const float wdecDen = 0.5f * (float)d * (float)(d + 1);

  for (int row = blockIdx.x * 8 + wv; row < Nrows; row += gridDim.x * 8) {
    const int b = row / W, w = row - b * W;
    const float* xp = x + ((size_t)b * T30 + (size_t)w * d) * 16 + c;

    float s = 0.f, sq = 0.f, dec = 0.f, first = 0.f, last = 0.f;
    for (int t = 0; t < d; t++) {
      float v = xp[t * 16];
      s += v; sq += v * v; dec += v * (float)(t + 1);
      if (t == 0) first = v;
      if (t == d - 1) last = v;
    }
    float mean = s * invd;
    float var  = fmaxf(sq * invd - mean * mean, 0.f);
    float sd   = sqrtf(var);
    float zs   = mean / sd;
    float rt   = last / first - 1.f;
    float dc   = dec / wdecDen;
    if (half == 0) stdbuf[wv][c] = sd;

    // centered values to LDS, layernorm across channels + depthwise conv
    float h1acc = 0.f;
    for (int t = 0; t < d; t++) {
      float v = xp[t * 16];
      if (half == 0) xcbuf[wv][t][c] = v - mean;
      float ls = v, lq = v * v;
#pragma unroll
      for (int off = 8; off; off >>= 1) { ls += __shfl_xor(ls, off, 16); lq += __shfl_xor(lq, off, 16); }
      float lm = ls * (1.f / 16.f);
      float lv = lq * (1.f / 16.f) - lm * lm;
      float xn = (v - lm) * rsqrtf(lv + EPSV) * lng + lnb;
      h1acc += xn * dw_k[c * d + t];
    }
    float h1v = h1acc + dwb;

    unsigned short* arow = A0 + (size_t)row * 352;
    if (half == 0) {
      h1raw[(size_t)row * 16 + c] = h1v;
      h1s += h1v; h1q += h1v * h1v;
      arow[240 + c] = f2bf(sd);
      arow[256 + c] = f2bf(zs);
      arow[272 + c] = f2bf(rt);
      arow[288 + c] = f2bf(dc);
      arow[304 + c] = f2bf(mean);
      arow[336 + c] = 0;   // K-pad zeros
      sS[0] += sd;  sQ[0] += sd * sd;
      sS[1] += zs;  sQ[1] += zs * zs;
      sS[2] += rt;  sQ[2] += rt * rt;
      sS[3] += dc;  sQ[3] += dc * dc;
      sS[4] += mean; sQ[4] += mean * mean;
    }

    // cov / corr over 120 upper-triangular pairs (all 32 lanes)
#pragma unroll
    for (int q = 0; q < 4; q++) {
      int p = lane + q * 32;
      if (p < 120) {
        int i = PAIRS.iu[p], j = PAIRS.ju[p];
        float sc = 0.f;
        for (int t = 0; t < d; t++) sc += xcbuf[wv][t][i] * xcbuf[wv][t][j];
        float cov = sc * invd;
        float corr = cov / (stdbuf[wv][i] * stdbuf[wv][j]);
        arow[p]       = f2bf(corr);
        arow[120 + p] = f2bf(cov);
        covS[q] += cov;  covQ[q] += cov * cov;
        corS[q] += corr; corQ[q] += corr * corr;
      }
    }
  }

  if (half == 0) {
    const int fbase[5] = {240, 256, 272, 288, 304};
#pragma unroll
    for (int k = 0; k < 5; k++) {
      atomicAdd(&featS[fbase[k] + c], sS[k]);
      atomicAdd(&featS[320 + fbase[k] + c], sQ[k]);
    }
    atomicAdd(&h1S[c], h1s); atomicAdd(&h1S[16 + c], h1q);
  }
#pragma unroll
  for (int q = 0; q < 4; q++) {
    int p = lane + q * 32;
    if (p < 120) {
      atomicAdd(&featS[p], corS[q]);        atomicAdd(&featS[320 + p], corQ[q]);
      atomicAdd(&featS[120 + p], covS[q]);  atomicAdd(&featS[320 + 120 + p], covQ[q]);
    }
  }
}

__global__ void finalize_kernel(const float* __restrict__ S1, const float* __restrict__ S2,
                                float invN, int nch, float* __restrict__ MI) {
  int i = blockIdx.x * blockDim.x + threadIdx.x;
  if (i < nch) {
    float m = S1[i] * invN;
    float v = S2[i] * invN - m * m;
    MI[i] = m;
    MI[nch + i] = rsqrtf(v + EPSV);
  }
}

// bn1+relu -> pointwise 16x16 -> stats of h2
__global__ void lf2_kernel(const float* __restrict__ h1raw, int Nrows,
                           const float* __restrict__ h1MI, const float* __restrict__ g1,
                           const float* __restrict__ b1, const float* __restrict__ pw_w,
                           const float* __restrict__ pw_b,
                           float* __restrict__ h2raw, float* __restrict__ h2S) {
  __shared__ float ls1[16], ls2[16];
  if (threadIdx.x < 16) { ls1[threadIdx.x] = 0.f; ls2[threadIdx.x] = 0.f; }
  __syncthreads();
  for (int row = blockIdx.x * blockDim.x + threadIdx.x; row < Nrows; row += gridDim.x * blockDim.x) {
    float hn[16];
#pragma unroll
    for (int cc = 0; cc < 16; cc++) {
      float v = h1raw[(size_t)row * 16 + cc];
      v = (v - h1MI[cc]) * h1MI[16 + cc] * g1[cc] + b1[cc];
      hn[cc] = fmaxf(v, 0.f);
    }
#pragma unroll
    for (int o = 0; o < 16; o++) {
      float acc = pw_b[o];
#pragma unroll
      for (int cc = 0; cc < 16; cc++) acc += pw_w[o * 16 + cc] * hn[cc];
      h2raw[(size_t)row * 16 + o] = acc;
      atomicAdd(&ls1[o], acc); atomicAdd(&ls2[o], acc * acc);
    }
  }
  __syncthreads();
  if (threadIdx.x < 16) {
    atomicAdd(&h2S[threadIdx.x], ls1[threadIdx.x]);
    atomicAdd(&h2S[16 + threadIdx.x], ls2[threadIdx.x]);
  }
}

// Normalize A0 in place (feat: bn only; lf cols 320..335: bn+relu from h2raw)
__global__ void normA0_kernel(unsigned short* __restrict__ A0, int Nrows,
                              const float* __restrict__ featMI,
                              const float* __restrict__ ext_g, const float* __restrict__ ext_b,
                              const float* __restrict__ h2raw, const float* __restrict__ h2MI,
                              const float* __restrict__ g2, const float* __restrict__ b2) {
  long gid = (long)blockIdx.x * blockDim.x + threadIdx.x;
  long row = gid / 11;
  int  seg = (int)(gid - row * 11);
  if (row >= Nrows) return;
  unsigned short* arow = A0 + (size_t)row * 352;
  if (seg < 10) {
    int f0 = seg * 32;
#pragma unroll
    for (int k = 0; k < 32; k++) {
      int f = f0 + k;
      float v = bf2f(arow[f]);
      v = (v - featMI[f]) * featMI[320 + f] * ext_g[f] + ext_b[f];
      arow[f] = f2bf(v);
    }
  } else {
#pragma unroll
    for (int cc = 0; cc < 16; cc++) {
      float v = h2raw[(size_t)row * 16 + cc];
      v = (v - h2MI[cc]) * h2MI[16 + cc] * g2[cc] + b2[cc];
      arow[320 + cc] = f2bf(fmaxf(v, 0.f));
    }
  }
}

// Swizzle f32 weight [Nout,K] into bf16 WMMA B fragments:
// Bfrag[((nt*ktiles+kt)*32+lane)*16+e] = W[n0+(lane&15)][k0+(lane>=16?8:0)+e+(e>=8?8:0)]
__global__ void prepB_kernel(const float* __restrict__ Wm, int Nout, int K,
                             int ktiles, int ntiles, unsigned short* __restrict__ Bfrag) {
  int total = ntiles * ktiles * 512;
  for (int gid = blockIdx.x * blockDim.x + threadIdx.x; gid < total; gid += gridDim.x * blockDim.x) {
    int e = gid & 15, lane = (gid >> 4) & 31, t = gid >> 9;
    int kt = t % ktiles, nt = t / ktiles;
    int n = nt * 16 + (lane & 15);
    int k = kt * 32 + ((lane >> 4) ? 8 : 0) + e + ((e >= 8) ? 8 : 0);
    float v = (n < Nout && k < K) ? Wm[n * K + k] : 0.f;
    Bfrag[gid] = f2bf(v);
  }
}

// bf16 GEMM via v_wmma_f32_16x16x32_bf16.
// B fragments async-staged to LDS; k-loop software pipelined; each wave
// computes two M-tiles sharing the B fragment. Fused bias + BN stats.
__global__ void gemm_kernel(const unsigned short* __restrict__ A, int lda,
                            const unsigned short* __restrict__ Bfrag,
                            const float* __restrict__ bias, int Nout,
                            unsigned short* __restrict__ Out, int ldo,
                            float* __restrict__ encS, int Nrows, int ktiles, int ntiles) {
  extern __shared__ unsigned short sB[];          // [ntiles*ktiles*512] bf16
  const int nfrag = ntiles * ktiles * 512;
  float* ls1 = (float*)(sB + nfrag);              // [176]
  float* ls2 = ls1 + 176;
  for (int i = threadIdx.x; i < 176; i += blockDim.x) { ls1[i] = 0.f; ls2[i] = 0.f; }

  // stage all B fragments into LDS with CDNA5 async copies (ASYNCcnt)
  {
    const int chunks = nfrag >> 3;                // 16B per chunk
    for (int i = threadIdx.x; i < chunks; i += blockDim.x) {
      unsigned lds = (unsigned)(uintptr_t)(&sB[i * 8]);
      async_g2l_b128(lds, Bfrag + i * 8);
    }
    wait_asynccnt0();
  }
  __syncthreads();

  const int lane = threadIdx.x & 31, wv = threadIdx.x >> 5;
  const int half = lane >> 4, lr = lane & 15;
  const int mpairs = Nrows >> 5;                  // Nrows always multiple of 32
  for (int mp = blockIdx.x * 8 + wv; mp < mpairs; mp += gridDim.x * 8) {
    const int m0 = mp << 5;
    const unsigned short* ar0 = A + (size_t)(m0 + lr) * lda + half * 8;
    const unsigned short* ar1 = ar0 + (size_t)16 * lda;
    for (int nt = 0; nt < ntiles; nt++) {
      v8f acc0 = {0.f,0.f,0.f,0.f,0.f,0.f,0.f,0.f};
      v8f acc1 = {0.f,0.f,0.f,0.f,0.f,0.f,0.f,0.f};
      const unsigned short* bp = sB + (size_t)nt * ktiles * 512 + lane * 16;
      v16bf a0 = load_afrag(ar0, ar0 + 16);
      v16bf a1 = load_afrag(ar1, ar1 + 16);
      v16bf bm = load_bfrag(bp);
      for (int kt = 0; kt < ktiles - 1; kt++) {
        // prefetch next k-tile before consuming current fragments
        const unsigned short* an0 = ar0 + (kt + 1) * 32;
        const unsigned short* an1 = ar1 + (kt + 1) * 32;
        v16bf a0n = load_afrag(an0, an0 + 16);
        v16bf a1n = load_afrag(an1, an1 + 16);
        v16bf bmn = load_bfrag(bp + (kt + 1) * 512);
        acc0 = __builtin_amdgcn_wmma_f32_16x16x32_bf16(false, a0, false, bm,
                                                       (short)0, acc0, false, false);
        acc1 = __builtin_amdgcn_wmma_f32_16x16x32_bf16(false, a1, false, bm,
                                                       (short)0, acc1, false, false);
        a0 = a0n; a1 = a1n; bm = bmn;
      }
      acc0 = __builtin_amdgcn_wmma_f32_16x16x32_bf16(false, a0, false, bm,
                                                     (short)0, acc0, false, false);
      acc1 = __builtin_amdgcn_wmma_f32_16x16x32_bf16(false, a1, false, bm,
                                                     (short)0, acc1, false, false);
      const int col = (nt << 4) + lr;
      const float bv = (col < Nout) ? bias[col] : 0.f;
      float s1 = 0.f, s2 = 0.f;
#pragma unroll
      for (int r = 0; r < 8; r++) {
        int ro = half ? 8 + r : r;
        float v0 = acc0[r] + bv;
        Out[(size_t)(m0 + ro) * ldo + col] = f2bf(v0);
        s1 += v0; s2 += v0 * v0;
        float v1 = acc1[r] + bv;
        Out[(size_t)(m0 + 16 + ro) * ldo + col] = f2bf(v1);
        s1 += v1; s2 += v1 * v1;
      }
      atomicAdd(&ls1[col], s1); atomicAdd(&ls2[col], s2);
    }
  }
  __syncthreads();
  const int npad = ntiles << 4;
  for (int i = threadIdx.x; i < npad; i += blockDim.x) {
    atomicAdd(&encS[i], ls1[i]); atomicAdd(&encS[npad + i], ls2[i]);
  }
}

// In-place BN + ReLU on bf16 activations, zero the K-pad tail.
__global__ void normrelu_kernel(unsigned short* __restrict__ A, int lda, int Nout,
                                const float* __restrict__ MI, const float* __restrict__ g,
                                const float* __restrict__ b, int Nrows) {
  for (int row = blockIdx.x * blockDim.x + threadIdx.x; row < Nrows; row += gridDim.x * blockDim.x) {
    unsigned short* p = A + (size_t)row * lda;
    for (int o = 0; o < Nout; o++) {
      float v = bf2f(p[o]);
      v = (v - MI[o]) * MI[Nout + o] * g[o] + b[o];
      p[o] = f2bf(fmaxf(v, 0.f));
    }
    for (int o = Nout; o < lda; o++) p[o] = 0;
  }
}

__global__ void initout_kernel(float* out, const float* lin_b, int B) {
  int i = blockIdx.x * blockDim.x + threadIdx.x;
  if (i < B) out[i] = lin_b[0];
}

// Final BN+ReLU of enc3 output folded into the 126->1 linear.
__global__ void final_kernel(const unsigned short* __restrict__ raw3, int W,
                             const float* __restrict__ MI, const float* __restrict__ g,
                             const float* __restrict__ b, const float* __restrict__ lin_w,
                             int base, float* __restrict__ out, int B) {
  int bi = blockIdx.x * blockDim.x + threadIdx.x;
  if (bi >= B) return;
  float acc = 0.f;
  for (int w = 0; w < W; w++) {
    const unsigned short* p = raw3 + (size_t)(bi * W + w) * 32;
#pragma unroll
    for (int o = 0; o < 21; o++) {
      float v = bf2f(p[o]);
      v = (v - MI[o]) * MI[21 + o] * g[o] + b[o];
      acc += fmaxf(v, 0.f) * lin_w[base + o * W + w];
    }
  }
  out[bi] += acc;
}

// ---------------------------------------------------------------------------
extern "C" void kernel_launch(void* const* d_in, const int* in_sizes, int n_in,
                              void* d_out, int out_size, void* d_ws, size_t ws_size,
                              hipStream_t stream) {
  (void)in_sizes; (void)n_in; (void)out_size; (void)ws_size;
  const float* x     = (const float*)d_in[0];
  const float* ln_g  = (const float*)d_in[1];
  const float* ln_b  = (const float*)d_in[2];
  const float* lin_w = (const float*)d_in[3];
  const float* lin_b = (const float*)d_in[4];

  char* wsb = (char*)d_ws;
  size_t off = 0;
  auto alloc = [&](size_t bytes) -> void* {
    void* p = wsb + off;
    off = (off + bytes + 255) & ~(size_t)255;
    return p;
  };
  unsigned short* A0 = (unsigned short*)alloc((size_t)NMAX * 352 * 2);
  unsigned short* A1 = (unsigned short*)alloc((size_t)NMAX * 192 * 2);
  unsigned short* A2 = (unsigned short*)alloc((size_t)NMAX * 96 * 2);
  unsigned short* A3 = (unsigned short*)alloc((size_t)NMAX * 64 * 2);
  unsigned short* raw3[3];
  raw3[0] = (unsigned short*)alloc((size_t)BATCH * 3 * 32 * 2);
  raw3[1] = (unsigned short*)alloc((size_t)BATCH * 2 * 32 * 2);
  raw3[2] = (unsigned short*)alloc((size_t)BATCH * 1 * 32 * 2);
  float* h1raw = (float*)alloc((size_t)NMAX * 16 * 4);
  float* h2raw = (float*)alloc((size_t)NMAX * 16 * 4);
  unsigned short* Bfrag = (unsigned short*)alloc(131072);
  float* statsArena = (float*)alloc(3 * 8192 * sizeof(float));

  zero_kernel<<<96, 256, 0, stream>>>(statsArena, 3 * 8192);

  const int dv[3] = {10, 15, 30};
  for (int di = 0; di < 3; di++) {
    const int d = dv[di], W = T30 / d, N = BATCH * W;
    const float invN = 1.0f / (float)N;
    void* const* leaf = d_in + 5 + di * 26;
    const float* ext_g = (const float*)leaf[0];
    const float* ext_b = (const float*)leaf[1];
    const float* dw_k  = (const float*)leaf[2];
    const float* dw_b  = (const float*)leaf[3];
    const float* g1    = (const float*)leaf[4];
    const float* b1    = (const float*)leaf[5];
    const float* pw_w  = (const float*)leaf[6];
    const float* pw_b  = (const float*)leaf[7];
    const float* g2    = (const float*)leaf[8];
    const float* b2    = (const float*)leaf[9];

    float* arena  = statsArena + di * 8192;
    float* featS  = arena;          float* featMI = arena + 640;
    float* h1S    = arena + 1280;   float* h1MI   = arena + 1312;
    float* h2S    = arena + 1344;   float* h2MI   = arena + 1376;

    extract_kernel<<<1024, 256, 0, stream>>>(x, ln_g, ln_b, dw_k, dw_b, d, W, N,
                                             A0, h1raw, featS, h1S);
    finalize_kernel<<<2, 256, 0, stream>>>(featS, featS + 320, invN, 320, featMI);
    finalize_kernel<<<1, 64, 0, stream>>>(h1S, h1S + 16, invN, 16, h1MI);
    lf2_kernel<<<1024, 256, 0, stream>>>(h1raw, N, h1MI, g1, b1, pw_w, pw_b, h2raw, h2S);
    finalize_kernel<<<1, 64, 0, stream>>>(h2S, h2S + 16, invN, 16, h2MI);
    {
      long total = (long)N * 11;
      int blocks = (int)((total + 255) / 256);
      normA0_kernel<<<blocks, 256, 0, stream>>>(A0, N, featMI, ext_g, ext_b,
                                                h2raw, h2MI, g2, b2);
    }

    unsigned short* Ain[4]  = {A0, A1, A2, A3};
    unsigned short* Aout[4] = {A1, A2, A3, raw3[di]};
    const int ldi[4] = {352, 192, 96, 64}, ldo[4] = {192, 96, 64, 32};
    const int Kd[4] = {336, 168, 84, 42}, No[4] = {168, 84, 42, 21};
    const int nt[4] = {11, 6, 3, 2}, kt[4] = {11, 6, 3, 2};
    for (int l = 0; l < 4; l++) {
      float* encS  = arena + 1408 + l * 352;
      float* encMI = arena + 2816 + l * 352;
      const float* Wl  = (const float*)leaf[10 + 4 * l];
      const float* bl  = (const float*)leaf[11 + 4 * l];
      const float* gl  = (const float*)leaf[12 + 4 * l];
      const float* bbl = (const float*)leaf[13 + 4 * l];
      prepB_kernel<<<128, 256, 0, stream>>>(Wl, No[l], Kd[l], kt[l], nt[l], Bfrag);
      const size_t shbytes = (size_t)nt[l] * kt[l] * 512 * 2 + 352 * 4;
      gemm_kernel<<<512, 256, shbytes, stream>>>(Ain[l], ldi[l], Bfrag, bl, No[l],
                                                 Aout[l], ldo[l], encS, N, kt[l], nt[l]);
      finalize_kernel<<<2, 256, 0, stream>>>(encS, encS + nt[l] * 16, invN, No[l], encMI);
      if (l < 3)
        normrelu_kernel<<<1024, 256, 0, stream>>>(Aout[l], ldo[l], No[l], encMI, gl, bbl, N);
    }
  }

  initout_kernel<<<256, 256, 0, stream>>>((float*)d_out, lin_b, BATCH);
  const int base[3] = {0, 63, 105}, Ws[3] = {3, 2, 1};
  for (int di = 0; di < 3; di++) {
    float* encMI3 = statsArena + di * 8192 + 2816 + 3 * 352;
    const float* g3  = (const float*)d_in[5 + di * 26 + 24];  // enc_g3
    const float* bb3 = (const float*)d_in[5 + di * 26 + 25];  // enc_bb3
    final_kernel<<<256, 256, 0, stream>>>(raw3[di], Ws[di], encMI3, g3, bb3,
                                          lin_w, base[di], (float*)d_out, BATCH);
  }
}
